// QuantMultiheadAttention_84138409329284
// MI455X (gfx1250) — 8-bit matrix operand order repaired and checked by reading, compiled, not run on hardware
//
#include <hip/hip_runtime.h>
#include <cstdint>
#include <cstddef>


typedef __attribute__((ext_vector_type(8))) int v8i;
typedef __attribute__((ext_vector_type(4))) int v4i;
typedef __attribute__((ext_vector_type(2))) int v2i;

constexpr int S_  = 2048;
constexpr int B_  = 2;
constexpr int E_  = 1024;
constexpr int H_  = 16;
constexpr int DH_ = 64;
constexpr int BH_ = B_ * H_;
constexpr int M_  = S_ * B_;

#define EPSQ 1e-12f

enum { SL_XQ = 0, SL_XK, SL_XV, SL_YQ, SL_YK, SL_YV, SL_SMAX, SL_PMAX, SL_ATTN, SL_YO, SL_COUNT };

__device__ __forceinline__ float slot_scale(const unsigned* sc, int slot) {
  return fmaxf(__uint_as_float(sc[slot]), EPSQ) * (1.0f / 127.0f);
}
__device__ __forceinline__ float clamp127(float q) {
  return fminf(fmaxf(q, -127.0f), 127.0f);
}

__device__ __forceinline__ v8i load_a8(const int8_t* __restrict__ A, int lda, int lane) {
  int row = lane & 15;
  int g8  = (lane >> 4) << 4;
  const int8_t* p = A + (size_t)row * lda + g8;
  v2i t0 = *(const v2i*)(p);
  v2i t1 = *(const v2i*)(p + 8);
  v2i t2 = *(const v2i*)(p + 32);
  v2i t3 = *(const v2i*)(p + 40);
  v8i a;
  a[0]=t0[0]; a[1]=t0[1]; a[2]=t1[0]; a[3]=t1[1];
  a[4]=t2[0]; a[5]=t2[1]; a[6]=t3[0]; a[7]=t3[1];
  return a;
}
__device__ __forceinline__ v8i load_b8(const int8_t* __restrict__ B, int ldb, int lane) {
  int col = lane & 15;
  int g16 = (lane >> 4) << 4;
  const int8_t* p = B + (size_t)col * ldb + g16;
  v4i u0 = *(const v4i*)(p);
  v4i u1 = *(const v4i*)(p + 32);
  v8i b;
  b[0]=u0[0]; b[1]=u0[1]; b[2]=u0[2]; b[3]=u0[3];
  b[4]=u1[0]; b[5]=u1[1]; b[6]=u1[2]; b[7]=u1[3];
  return b;
}
__device__ __forceinline__ v8i wmma_i8(v8i a, v8i b, v8i c) {
  return __builtin_amdgcn_wmma_i32_16x16x64_iu8(true, a, true, b, c, false, false);
}

__global__ void k_init_scalars(unsigned* sc) {
  if (threadIdx.x < 64) sc[threadIdx.x] = 0u;
}

__global__ void k_amax(const float* __restrict__ x, size_t n, unsigned* __restrict__ out) {
  __shared__ float red[256];
  float lmax = 0.0f;
  for (size_t i = (size_t)blockIdx.x * blockDim.x + threadIdx.x; i < n;
       i += (size_t)gridDim.x * blockDim.x)
    lmax = fmaxf(lmax, fabsf(x[i]));
  red[threadIdx.x] = lmax; __syncthreads();
  for (int s = 128; s; s >>= 1) {
    if (threadIdx.x < s) red[threadIdx.x] = fmaxf(red[threadIdx.x], red[threadIdx.x + s]);
    __syncthreads();
  }
  if (threadIdx.x == 0) atomicMax(out, __float_as_uint(red[0]));
}

__global__ void k_quant_tensor(const float* __restrict__ x, size_t n,
                               const unsigned* __restrict__ sc, int slot,
                               int8_t* __restrict__ out) {
  float inv = 1.0f / slot_scale(sc, slot);
  for (size_t i = (size_t)blockIdx.x * blockDim.x + threadIdx.x; i < n;
       i += (size_t)gridDim.x * blockDim.x)
    out[i] = (int8_t)clamp127(rintf(x[i] * inv));
}

__global__ void k_quant_w(const float* __restrict__ w, int8_t* __restrict__ w8,
                          float* __restrict__ rscale, int K) {
  __shared__ float red[256];
  int row = blockIdx.x;
  const float* wr = w + (size_t)row * K;
  float lmax = 0.0f;
  for (int i = threadIdx.x; i < K; i += blockDim.x) lmax = fmaxf(lmax, fabsf(wr[i]));
  red[threadIdx.x] = lmax; __syncthreads();
  for (int s = 128; s; s >>= 1) {
    if (threadIdx.x < s) red[threadIdx.x] = fmaxf(red[threadIdx.x], red[threadIdx.x + s]);
    __syncthreads();
  }
  float scale = fmaxf(red[0], EPSQ) * (1.0f / 127.0f);
  if (threadIdx.x == 0) rscale[row] = scale;
  float inv = 1.0f / scale;
  for (int i = threadIdx.x; i < K; i += blockDim.x)
    w8[(size_t)row * K + i] = (int8_t)clamp127(rintf(wr[i] * inv));
}

__global__ void k_quant_head(const float* __restrict__ y, const unsigned* __restrict__ sc,
                             int slot, int8_t* __restrict__ out, int transp) {
  size_t n = (size_t)M_ * E_;
  float inv = 1.0f / slot_scale(sc, slot);
  for (size_t i = (size_t)blockIdx.x * blockDim.x + threadIdx.x; i < n;
       i += (size_t)gridDim.x * blockDim.x) {
    size_t s = i / (size_t)(B_ * E_);
    int rem = (int)(i % (size_t)(B_ * E_));
    int b = rem / E_, e = rem % E_;
    int h = e >> 6, d = e & 63;
    int j = b * H_ + h;
    int8_t q = (int8_t)clamp127(rintf(y[i] * inv));
    if (transp) out[((size_t)j * DH_ + d) * S_ + s] = q;
    else        out[((size_t)j * S_  + s) * DH_ + d] = q;
  }
}

__global__ void k_final_fq(const float* __restrict__ y, size_t n,
                           const unsigned* __restrict__ sc, int slot,
                           float* __restrict__ out) {
  float scale = slot_scale(sc, slot);
  float inv = 1.0f / scale;
  for (size_t i = (size_t)blockIdx.x * blockDim.x + threadIdx.x; i < n;
       i += (size_t)gridDim.x * blockDim.x)
    out[i] = clamp127(rintf(y[i] * inv)) * scale;
}


__global__ void __launch_bounds__(128)
gemm_i8_proj(const int8_t* __restrict__ A, const int8_t* __restrict__ Bw,
             const float* __restrict__ rscale, const float* __restrict__ bias,
             const unsigned* __restrict__ sc, int slotA,
             float* __restrict__ Y, unsigned* __restrict__ amaxY,
             int Kk, int Nn) {
  __shared__ float red[128];
  int lane = threadIdx.x & 31;
  int wave = threadIdx.x >> 5;
  int tn2 = blockIdx.x * 4 + wave;
  int tm  = blockIdx.y;
  const int8_t* Ab  = A  + (size_t)(tm * 16) * Kk;
  const int8_t* Bb0 = Bw + (size_t)(tn2 * 32) * Kk;
  const int8_t* Bb1 = Bb0 + (size_t)16 * Kk;
  v8i acc0 = {}, acc1 = {};
  v8i a  = load_a8(Ab,  Kk, lane);
  v8i b0 = load_b8(Bb0, Kk, lane);
  v8i b1 = load_b8(Bb1, Kk, lane);
  for (int k0 = 64; k0 < Kk; k0 += 64) {
    __builtin_prefetch(Ab  + k0 + 64, 0, 3);
    __builtin_prefetch(Bb0 + k0 + 64, 0, 3);
    v8i an  = load_a8(Ab  + k0, Kk, lane);
    v8i bn0 = load_b8(Bb0 + k0, Kk, lane);
    v8i bn1 = load_b8(Bb1 + k0, Kk, lane);
    acc0 = wmma_i8(a, b0, acc0);
    acc1 = wmma_i8(a, b1, acc1);
    a = an; b0 = bn0; b1 = bn1;
  }
  acc0 = wmma_i8(a, b0, acc0);
  acc1 = wmma_i8(a, b1, acc1);

  float sa = slot_scale(sc, slotA);
  int n0 = tn2 * 32 + (lane & 15);
  int n1 = n0 + 16;
  float s0 = sa * rscale[n0], s1 = sa * rscale[n1];
  float bn0f = bias[n0], bn1f = bias[n1];
  int rb = tm * 16 + ((lane >> 4) << 3);
  float lmax = 0.0f;
#pragma unroll
  for (int r = 0; r < 8; r++) {
    float y0 = (float)acc0[r] * s0 + bn0f;
    float y1 = (float)acc1[r] * s1 + bn1f;
    Y[(size_t)(rb + r) * Nn + n0] = y0;
    Y[(size_t)(rb + r) * Nn + n1] = y1;
    lmax = fmaxf(lmax, fmaxf(fabsf(y0), fabsf(y1)));
  }
  red[threadIdx.x] = lmax; __syncthreads();
  for (int t = 64; t; t >>= 1) {
    if (threadIdx.x < t) red[threadIdx.x] = fmaxf(red[threadIdx.x], red[threadIdx.x + t]);
    __syncthreads();
  }
  if (threadIdx.x == 0) atomicMax(amaxY, __float_as_uint(red[0]));
}

__global__ void __launch_bounds__(128)
gemm_scores(const int8_t* __restrict__ q8, const int8_t* __restrict__ k8,
            int* __restrict__ scores, unsigned* __restrict__ smax) {
  __shared__ float red[128];
  int lane = threadIdx.x & 31;
  int wave = threadIdx.x >> 5;
  int h = blockIdx.z;
  int tn2 = blockIdx.x * 4 + wave;
  int tm  = blockIdx.y;
  const int8_t* A   = q8 + ((size_t)h * S_ + tm * 16) * DH_;
  const int8_t* Bm0 = k8 + ((size_t)h * S_ + tn2 * 32) * DH_;
  const int8_t* Bm1 = Bm0 + (size_t)16 * DH_;
  v8i a  = load_a8(A,   DH_, lane);
  v8i b0 = load_b8(Bm0, DH_, lane);
  v8i b1 = load_b8(Bm1, DH_, lane);
  v8i acc0 = {}, acc1 = {};
  acc0 = wmma_i8(a, b0, acc0);
  acc1 = wmma_i8(a, b1, acc1);

  int n0 = tn2 * 32 + (lane & 15);
  int rb = tm * 16 + ((lane >> 4) << 3);
  int* out = scores + (size_t)h * S_ * S_;
  float lmax = 0.0f;
#pragma unroll
  for (int r = 0; r < 8; r++) {
    int v0 = acc0[r], v1 = acc1[r];
    out[(size_t)(rb + r) * S_ + n0]      = v0;
    out[(size_t)(rb + r) * S_ + n0 + 16] = v1;
    lmax = fmaxf(lmax, fmaxf(fabsf((float)v0), fabsf((float)v1)));
  }
  red[threadIdx.x] = lmax; __syncthreads();
  for (int t = 64; t; t >>= 1) {
    if (threadIdx.x < t) red[threadIdx.x] = fmaxf(red[threadIdx.x], red[threadIdx.x + t]);
    __syncthreads();
  }
  if (threadIdx.x == 0) atomicMax(smax, __float_as_uint(red[0]));
}

__device__ __forceinline__ float score_x(int v, float c, float ss8) {
  return clamp127(rintf((float)v * c)) * ss8;
}

__global__ void k_softmax_stats(const int* __restrict__ scores, const unsigned* __restrict__ sc,
                                float* __restrict__ mrow, float* __restrict__ zrow,
                                unsigned* __restrict__ pmax) {
  __shared__ float red[256];
  size_t row = blockIdx.x;
  const int* sr = scores + row * (size_t)S_;
  float maxi = fmaxf(__uint_as_float(sc[SL_SMAX]), 1.0f);
  float c = 127.0f / maxi;
  float sq = slot_scale(sc, SL_YQ);
  float sk = slot_scale(sc, SL_YK);
  float ss8 = sq * sk * maxi * (1.0f / 127.0f) * 0.125f;
  float m = -1e30f;
  for (int i = threadIdx.x; i < S_; i += blockDim.x)
    m = fmaxf(m, score_x(sr[i], c, ss8));
  red[threadIdx.x] = m; __syncthreads();
  for (int s = 128; s; s >>= 1) {
    if (threadIdx.x < s) red[threadIdx.x] = fmaxf(red[threadIdx.x], red[threadIdx.x + s]);
    __syncthreads();
  }
  m = red[0];
  __syncthreads();
  float z = 0.0f;
  for (int i = threadIdx.x; i < S_; i += blockDim.x)
    z += expf(score_x(sr[i], c, ss8) - m);
  red[threadIdx.x] = z; __syncthreads();
  for (int s = 128; s; s >>= 1) {
    if (threadIdx.x < s) red[threadIdx.x] += red[threadIdx.x + s];
    __syncthreads();
  }
  if (threadIdx.x == 0) {
    mrow[row] = m;
    zrow[row] = red[0];
    atomicMax(pmax, __float_as_uint(1.0f / red[0]));
  }
}

__global__ void k_quant_probs(const int* __restrict__ scores, const float* __restrict__ mrow,
                              const float* __restrict__ zrow, const unsigned* __restrict__ sc,
                              int8_t* __restrict__ probs) {
  size_t n = (size_t)BH_ * S_ * S_;
  float maxi = fmaxf(__uint_as_float(sc[SL_SMAX]), 1.0f);
  float c = 127.0f / maxi;
  float sq = slot_scale(sc, SL_YQ);
  float sk = slot_scale(sc, SL_YK);
  float ss8 = sq * sk * maxi * (1.0f / 127.0f) * 0.125f;
  float pinv = 1.0f / slot_scale(sc, SL_PMAX);
  for (size_t i = (size_t)blockIdx.x * blockDim.x + threadIdx.x; i < n;
       i += (size_t)gridDim.x * blockDim.x) {
    size_t row = i >> 11;
    float x = score_x(scores[i], c, ss8);
    float p = expf(x - mrow[row]) / zrow[row];
    probs[i] = (int8_t)clamp127(rintf(p * pinv));
  }
}

__global__ void __launch_bounds__(128)
gemm_attn(const int8_t* __restrict__ p8, const int8_t* __restrict__ v8t,
          const unsigned* __restrict__ sc, float* __restrict__ attn) {
  int lane = threadIdx.x & 31;
  int wave = threadIdx.x >> 5;
  int h = blockIdx.z;
  int tm  = blockIdx.y * 2 + (wave >> 1);
  int tnp = wave & 1;
  const int8_t* A   = p8  + (size_t)h * S_ * S_ + (size_t)(tm * 16) * S_;
  const int8_t* Bm0 = v8t + (size_t)h * DH_ * S_ + (size_t)(tnp * 32) * S_;
  const int8_t* Bm1 = Bm0 + (size_t)16 * S_;
  v8i acc0 = {}, acc1 = {};
  v8i a  = load_a8(A,   S_, lane);
  v8i b0 = load_b8(Bm0, S_, lane);
  v8i b1 = load_b8(Bm1, S_, lane);
  for (int k0 = 64; k0 < S_; k0 += 64) {
    __builtin_prefetch(A   + k0 + 64, 0, 3);
    __builtin_prefetch(Bm0 + k0 + 64, 0, 3);
    v8i an  = load_a8(A   + k0, S_, lane);
    v8i bn0 = load_b8(Bm0 + k0, S_, lane);
    v8i bn1 = load_b8(Bm1 + k0, S_, lane);
    acc0 = wmma_i8(a, b0, acc0);
    acc1 = wmma_i8(a, b1, acc1);
    a = an; b0 = bn0; b1 = bn1;
  }
  acc0 = wmma_i8(a, b0, acc0);
  acc1 = wmma_i8(a, b1, acc1);

  float s = slot_scale(sc, SL_PMAX) * slot_scale(sc, SL_YV);
  int d0 = tnp * 32 + (lane & 15);
  int b = h >> 4, hh = h & 15;
  int rb = tm * 16 + ((lane >> 4) << 3);
#pragma unroll
  for (int r = 0; r < 8; r++) {
    int t = rb + r;
    float* dst = attn + ((size_t)t * B_ + b) * E_ + hh * DH_;
    dst[d0]      = (float)acc0[r] * s;
    dst[d0 + 16] = (float)acc1[r] * s;
  }
}

extern "C" void kernel_launch(void* const* d_in, const int* in_sizes, int n_in,
                              void* d_out, int out_size, void* d_ws, size_t ws_size,
                              hipStream_t stream) {
  (void)in_sizes; (void)n_in; (void)out_size; (void)ws_size;

  const float* query = (const float*)d_in[0];
  const float* key   = (const float*)d_in[1];
  const float* value = (const float*)d_in[2];
  const float* W[4]  = { (const float*)d_in[3], (const float*)d_in[5],
                         (const float*)d_in[7], (const float*)d_in[9] };
  const float* bias[4] = { (const float*)d_in[4], (const float*)d_in[6],
                           (const float*)d_in[8], (const float*)d_in[10] };
  float* out = (float*)d_out;

  char* base = (char*)d_ws;
  size_t off = 0;
  auto carve = [&](size_t bytes) -> char* {
    char* p = base + off;
    off += (bytes + 255) & ~(size_t)255;
    return p;
  };
  unsigned* SC   = (unsigned*)carve(256);
  float*  RS     = (float*)  carve(4 * E_ * sizeof(float));
  int8_t* AI8    = (int8_t*) carve((size_t)M_ * E_);
  int8_t* WI8    = (int8_t*) carve((size_t)4 * E_ * E_);
  float*  YBUF   = (float*)  carve((size_t)M_ * E_ * 4);
  float*  ATTN   = (float*)  carve((size_t)M_ * E_ * 4);
  int8_t* QI8    = (int8_t*) carve((size_t)BH_ * S_ * DH_);
  int8_t* KI8    = (int8_t*) carve((size_t)BH_ * S_ * DH_);
  int8_t* VI8T   = (int8_t*) carve((size_t)BH_ * DH_ * S_);
  float*  MROW   = (float*)  carve((size_t)BH_ * S_ * 4);
  float*  ZROW   = (float*)  carve((size_t)BH_ * S_ * 4);
  int8_t* PROBS  = (int8_t*) carve((size_t)BH_ * S_ * S_);
  int*    SCORES = (int*)    carve((size_t)BH_ * S_ * S_ * 4);

  const size_t NTENS = (size_t)M_ * E_;
  dim3 gProj(E_ / 128, M_ / 16);
  dim3 gScores(S_ / 128, S_ / 16, BH_);
  dim3 gAttn(1, S_ / 32, BH_);

  k_init_scalars<<<1, 64, 0, stream>>>(SC);

  for (int i = 0; i < 4; i++)
    k_quant_w<<<E_, 256, 0, stream>>>(W[i], WI8 + (size_t)i * E_ * E_, RS + (size_t)i * E_, E_);

  const float* xin[3]  = { query, key, value };
  const int   slotX[3] = { SL_XQ, SL_XK, SL_XV };
  const int   slotY[3] = { SL_YQ, SL_YK, SL_YV };
  int8_t* head8[3] = { QI8, KI8, VI8T };
  for (int i = 0; i < 3; i++) {
    k_amax<<<2048, 256, 0, stream>>>(xin[i], NTENS, SC + slotX[i]);
    k_quant_tensor<<<4096, 256, 0, stream>>>(xin[i], NTENS, SC, slotX[i], AI8);
    gemm_i8_proj<<<gProj, 128, 0, stream>>>(AI8, WI8 + (size_t)i * E_ * E_, RS + (size_t)i * E_,
                                            bias[i], SC, slotX[i], YBUF, SC + slotY[i], E_, E_);
    k_quant_head<<<4096, 256, 0, stream>>>(YBUF, SC, slotY[i], head8[i], i == 2 ? 1 : 0);
  }

  gemm_scores<<<gScores, 128, 0, stream>>>(QI8, KI8, SCORES, SC + SL_SMAX);
  k_softmax_stats<<<BH_ * S_, 256, 0, stream>>>(SCORES, SC, MROW, ZROW, SC + SL_PMAX);
  k_quant_probs<<<32768, 256, 0, stream>>>(SCORES, MROW, ZROW, SC, PROBS);
  gemm_attn<<<gAttn, 128, 0, stream>>>(PROBS, VI8T, SC, ATTN);

  k_amax<<<2048, 256, 0, stream>>>(ATTN, NTENS, SC + SL_ATTN);
  k_quant_tensor<<<4096, 256, 0, stream>>>(ATTN, NTENS, SC, SL_ATTN, AI8);
  gemm_i8_proj<<<gProj, 128, 0, stream>>>(AI8, WI8 + (size_t)3 * E_ * E_, RS + (size_t)3 * E_,
                                          bias[3], SC, SL_ATTN, YBUF, SC + SL_YO, E_, E_);
  k_final_fq<<<4096, 256, 0, stream>>>(YBUF, NTENS, SC, SL_YO, out);
}
